// LiquidNeuronCPU_18769007084266
// MI455X (gfx1250) — compile-verified, matrix operand order checked by reading, not run on hardware
//
#include <hip/hip_runtime.h>

// ---------------------------------------------------------------------------
// LiquidNeuron fused pipeline for gfx1250 (MI455X).
//   out = 5*tanh( LayerNorm( clip( clip(x@Ws^T,±10) + x@(Wm+Wf)^T, ±10) )/5 )
// GEMMs: bf16 WMMA (v_wmma_f32_16x16x32_bf16), f32 accumulate, double-buffered
// LDS tiles filled with CDNA5 async global->LDS copies (ASYNCcnt).
// ---------------------------------------------------------------------------

typedef __attribute__((ext_vector_type(16))) __bf16 v16bf;
typedef __attribute__((ext_vector_type(8)))  float  v8f;
typedef int v4i_ __attribute__((vector_size(16)));   // matches builtin proto

#define BROWS 4096
#define DIM   2048

// LDS tile geometry (elements). Row stride 72 (=36 words) makes the 16 lane
// base banks {r*36 mod 64} all distinct -> conflict-free ds_load_b128.
#define ASTR    72
#define A_ELEMS (128 * ASTR)          // 9216
#define B_ELEMS (64 * ASTR)           // 4608
#define BUF_ELEMS (A_ELEMS + 2 * B_ELEMS)  // 18432 elems = 36 KB per buffer

#if __has_builtin(__builtin_amdgcn_global_load_async_to_lds_b128)
#define ASYNC_COPY16(dst, src)                                             \
  __builtin_amdgcn_global_load_async_to_lds_b128(                          \
      (__attribute__((address_space(1))) v4i_*)(src),                      \
      (__attribute__((address_space(3))) v4i_*)(dst), 0, 0)
#if __has_builtin(__builtin_amdgcn_s_wait_asynccnt)
#define ASYNC_WAIT() __builtin_amdgcn_s_wait_asynccnt(0)
#else
#define ASYNC_WAIT() asm volatile("s_wait_asynccnt 0x0" ::: "memory")
#endif
#else
// Fallback: stage through VGPRs with plain loads + ds stores.
#define ASYNC_COPY16(dst, src)                                             \
  do { *(uint4*)(dst) = *(const uint4*)(src); } while (0)
#define ASYNC_WAIT() do {} while (0)
#endif

struct bf16x4 { __bf16 v[4]; };

// ---- pack x: clip to ±5, convert fp32 -> bf16 -----------------------------
__global__ __launch_bounds__(256) void ln_pack_x(const float* __restrict__ x,
                                                 __bf16* __restrict__ xb) {
  int i = (blockIdx.x * 256 + threadIdx.x) * 4;
  float4 f = *(const float4*)(x + i);
  bf16x4 o;
  o.v[0] = (__bf16)fminf(fmaxf(f.x, -5.f), 5.f);
  o.v[1] = (__bf16)fminf(fmaxf(f.y, -5.f), 5.f);
  o.v[2] = (__bf16)fminf(fmaxf(f.z, -5.f), 5.f);
  o.v[3] = (__bf16)fminf(fmaxf(f.w, -5.f), 5.f);
  *(bf16x4*)(xb + i) = o;
}

// ---- pack weights: Ws -> bf16, (Wm+Wf) -> bf16 ----------------------------
__global__ __launch_bounds__(256) void ln_pack_w(const float* __restrict__ wslow,
                                                 const float* __restrict__ wmed,
                                                 const float* __restrict__ wfast,
                                                 __bf16* __restrict__ wsb,
                                                 __bf16* __restrict__ wmfb) {
  int i = (blockIdx.x * 256 + threadIdx.x) * 4;
  float4 s = *(const float4*)(wslow + i);
  float4 m = *(const float4*)(wmed + i);
  float4 f = *(const float4*)(wfast + i);
  bf16x4 os, omf;
  os.v[0] = (__bf16)s.x;  omf.v[0] = (__bf16)(m.x + f.x);
  os.v[1] = (__bf16)s.y;  omf.v[1] = (__bf16)(m.y + f.y);
  os.v[2] = (__bf16)s.z;  omf.v[2] = (__bf16)(m.z + f.z);
  os.v[3] = (__bf16)s.w;  omf.v[3] = (__bf16)(m.w + f.w);
  *(bf16x4*)(wsb + i)  = os;
  *(bf16x4*)(wmfb + i) = omf;
}

// 32-byte LDS fragment read as 2 x 16-byte (16B-aligned) pieces.
static __device__ inline v16bf ld32(const __bf16* p) {
  v16bf r;
  ((uint4*)&r)[0] = ((const uint4*)p)[0];
  ((uint4*)&r)[1] = ((const uint4*)(p + 8))[0];
  return r;
}

// ---- dual-weight WMMA GEMM, LDS double-buffered ---------------------------
// Block tile 128(M) x 64(N), K-panel 64. 8 waves, each owning a 32x32 output
// patch with separate slow / (medium+fast) accumulators so the per-path clip
// is applied in the epilogue.
//
// Fragment note: for 16-bit WMMA the A(16x32) and B(32x16) fragments share the
// same (lane-half, K-slot) placement, so a K-permutation applied identically
// to A and B leaves the dot product unchanged: lane L (r=L&15, h=L>>4) reads
// 16 contiguous bf16 from row r at K-offset h*16.
__global__ __launch_bounds__(256) void ln_gemm_dual(const __bf16* __restrict__ xb,
                                                    const __bf16* __restrict__ wsb,
                                                    const __bf16* __restrict__ wmfb,
                                                    float* __restrict__ pre) {
  __shared__ __align__(16) __bf16 lds[2][BUF_ELEMS];

  const int tid  = threadIdx.x;
  const int lane = tid & 31;
  const int wid  = tid >> 5;
  const int wm   = wid & 3;       // 0..3  -> M sub-tile
  const int wn   = wid >> 2;      // 0..1  -> N sub-tile
  const int r    = lane & 15;
  const int h    = lane >> 4;

  const int Mb0 = blockIdx.y * 128;
  const int Nb0 = blockIdx.x * 64;

  // LDS fragment offsets (elements), constant over the K loop.
  const int aoff0 = (wm * 32 + r) * ASTR + h * 16;
  const int aoff1 = (wm * 32 + 16 + r) * ASTR + h * 16;
  const int boff0 = (wn * 32 + r) * ASTR + h * 16;
  const int boff1 = (wn * 32 + 16 + r) * ASTR + h * 16;

  // Per-thread staging assignments: each chunk is 16 bytes (8 bf16).
  const int arow[4] = {(tid + 0) >> 3, (tid + 256) >> 3, (tid + 512) >> 3,
                       (tid + 768) >> 3};
  const int acol = (tid & 7) * 8;
  const int brow[2] = {(tid + 0) >> 3, (tid + 256) >> 3};

  v8f accS[2][2], accF[2][2];
  const v8f vz = {0.f, 0.f, 0.f, 0.f, 0.f, 0.f, 0.f, 0.f};
  accS[0][0] = vz; accS[0][1] = vz; accS[1][0] = vz; accS[1][1] = vz;
  accF[0][0] = vz; accF[0][1] = vz; accF[1][0] = vz; accF[1][1] = vz;

  // Prologue: fill buffer 0 with K-panel [0,64).
  {
    __bf16* La = &lds[0][0];
    __bf16* Lb = &lds[0][A_ELEMS];
    __bf16* Lc = &lds[0][A_ELEMS + B_ELEMS];
#pragma unroll
    for (int t = 0; t < 4; ++t)
      ASYNC_COPY16(&La[arow[t] * ASTR + acol],
                   xb + (size_t)(Mb0 + arow[t]) * DIM + acol);
#pragma unroll
    for (int t = 0; t < 2; ++t) {
      ASYNC_COPY16(&Lb[brow[t] * ASTR + acol],
                   wsb + (size_t)(Nb0 + brow[t]) * DIM + acol);
      ASYNC_COPY16(&Lc[brow[t] * ASTR + acol],
                   wmfb + (size_t)(Nb0 + brow[t]) * DIM + acol);
    }
  }
  ASYNC_WAIT();
  __syncthreads();

  int cur = 0;
  for (int k0 = 0; k0 < DIM; k0 += 64) {
    // Stage next K-panel into the other buffer while we compute on `cur`.
    if (k0 + 64 < DIM) {
      const int kn = k0 + 64;
      __bf16* La = &lds[cur ^ 1][0];
      __bf16* Lb = &lds[cur ^ 1][A_ELEMS];
      __bf16* Lc = &lds[cur ^ 1][A_ELEMS + B_ELEMS];
#pragma unroll
      for (int t = 0; t < 4; ++t)
        ASYNC_COPY16(&La[arow[t] * ASTR + acol],
                     xb + (size_t)(Mb0 + arow[t]) * DIM + kn + acol);
#pragma unroll
      for (int t = 0; t < 2; ++t) {
        ASYNC_COPY16(&Lb[brow[t] * ASTR + acol],
                     wsb + (size_t)(Nb0 + brow[t]) * DIM + kn + acol);
        ASYNC_COPY16(&Lc[brow[t] * ASTR + acol],
                     wmfb + (size_t)(Nb0 + brow[t]) * DIM + kn + acol);
      }
    }

    // Consume current buffer: 2 k-substeps of 32, 8 WMMAs each.
    const __bf16* base = &lds[cur][0];
#pragma unroll
    for (int ks = 0; ks < 64; ks += 32) {
      v16bf a0 = ld32(base + aoff0 + ks);
      v16bf a1 = ld32(base + aoff1 + ks);
      v16bf s0 = ld32(base + A_ELEMS + boff0 + ks);
      v16bf s1 = ld32(base + A_ELEMS + boff1 + ks);
      v16bf f0 = ld32(base + A_ELEMS + B_ELEMS + boff0 + ks);
      v16bf f1 = ld32(base + A_ELEMS + B_ELEMS + boff1 + ks);

      accS[0][0] = __builtin_amdgcn_wmma_f32_16x16x32_bf16(false, a0, false, s0,
                     (short)0, accS[0][0], false, false);
      accS[0][1] = __builtin_amdgcn_wmma_f32_16x16x32_bf16(false, a0, false, s1,
                     (short)0, accS[0][1], false, false);
      accS[1][0] = __builtin_amdgcn_wmma_f32_16x16x32_bf16(false, a1, false, s0,
                     (short)0, accS[1][0], false, false);
      accS[1][1] = __builtin_amdgcn_wmma_f32_16x16x32_bf16(false, a1, false, s1,
                     (short)0, accS[1][1], false, false);
      accF[0][0] = __builtin_amdgcn_wmma_f32_16x16x32_bf16(false, a0, false, f0,
                     (short)0, accF[0][0], false, false);
      accF[0][1] = __builtin_amdgcn_wmma_f32_16x16x32_bf16(false, a0, false, f1,
                     (short)0, accF[0][1], false, false);
      accF[1][0] = __builtin_amdgcn_wmma_f32_16x16x32_bf16(false, a1, false, f0,
                     (short)0, accF[1][0], false, false);
      accF[1][1] = __builtin_amdgcn_wmma_f32_16x16x32_bf16(false, a1, false, f1,
                     (short)0, accF[1][1], false, false);
    }

    ASYNC_WAIT();      // next buffer fully written (per-wave)
    __syncthreads();   // everyone done reading `cur` and done filling `next`
    cur ^= 1;
  }

  // Epilogue: per-path clips, write pre_act fp32.
  // C/D layout: VGPR j of lane (r,h) holds element (M = j + 8*h, N = r).
#pragma unroll
  for (int mi = 0; mi < 2; ++mi) {
#pragma unroll
    for (int ni = 0; ni < 2; ++ni) {
      const int row0 = Mb0 + wm * 32 + mi * 16 + 8 * h;
      const int col  = Nb0 + wn * 32 + ni * 16 + r;
#pragma unroll
      for (int j = 0; j < 8; ++j) {
        float sv = fminf(fmaxf(accS[mi][ni][j], -10.f), 10.f);
        float pv = fminf(fmaxf(sv + accF[mi][ni][j], -10.f), 10.f);
        pre[(size_t)(row0 + j) * DIM + col] = pv;
      }
    }
  }
}

// ---- per-row LayerNorm + soft clamp --------------------------------------
__global__ __launch_bounds__(256) void ln_norm_tanh(const float* __restrict__ pre,
                                                    const float* __restrict__ gamma,
                                                    const float* __restrict__ beta,
                                                    float* __restrict__ out) {
  __shared__ float s1[256];
  __shared__ float s2[256];
  const float* p = pre + (size_t)blockIdx.x * DIM;
  float v[8];
  float sum = 0.f, sq = 0.f;
#pragma unroll
  for (int j = 0; j < 8; ++j) {
    float t = p[threadIdx.x + j * 256];
    v[j] = t; sum += t; sq += t * t;
  }
  s1[threadIdx.x] = sum;
  s2[threadIdx.x] = sq;
  __syncthreads();
  for (int off = 128; off > 0; off >>= 1) {
    if (threadIdx.x < off) {
      s1[threadIdx.x] += s1[threadIdx.x + off];
      s2[threadIdx.x] += s2[threadIdx.x + off];
    }
    __syncthreads();
  }
  const float mean = s1[0] * (1.f / DIM);
  const float var  = s2[0] * (1.f / DIM) - mean * mean;
  const float rs   = rsqrtf(var + 1e-5f);
  float* o = out + (size_t)blockIdx.x * DIM;
#pragma unroll
  for (int j = 0; j < 8; ++j) {
    int c = threadIdx.x + j * 256;
    float n = (v[j] - mean) * rs * gamma[c] + beta[c];
    o[c] = 5.f * tanhf(n * 0.2f);
  }
}

extern "C" void kernel_launch(void* const* d_in, const int* in_sizes, int n_in,
                              void* d_out, int out_size, void* d_ws, size_t ws_size,
                              hipStream_t stream) {
  const float* x     = (const float*)d_in[0];
  const float* wslow = (const float*)d_in[1];
  const float* wmed  = (const float*)d_in[2];
  const float* wfast = (const float*)d_in[3];
  const float* gamma = (const float*)d_in[4];
  const float* beta  = (const float*)d_in[5];
  float* out = (float*)d_out;

  char* ws = (char*)d_ws;
  __bf16* xb   = (__bf16*)(ws);                         // 16 MB
  __bf16* wsb  = (__bf16*)(ws + (size_t)(16u << 20));   //  8 MB
  __bf16* wmfb = (__bf16*)(ws + (size_t)(24u << 20));   //  8 MB
  float*  pre  = (float*) (ws + (size_t)(32u << 20));   // 32 MB

  ln_pack_x<<<(BROWS * DIM) / 1024, 256, 0, stream>>>(x, xb);
  ln_pack_w<<<(DIM * DIM) / 1024, 256, 0, stream>>>(wslow, wmed, wfast, wsb, wmfb);
  ln_gemm_dual<<<dim3(DIM / 64, BROWS / 128), 256, 0, stream>>>(xb, wsb, wmfb, pre);
  ln_norm_tanh<<<BROWS, 256, 0, stream>>>(pre, gamma, beta, out);
}
